// RippleLinear_9363028705494
// MI455X (gfx1250) — matrix operand order repaired and checked by reading, not compiled, not run on hardware
//
#include <hip/hip_runtime.h>
#include <hip/hip_bf16.h>

typedef __attribute__((ext_vector_type(16))) _Float16 v16h;
typedef __attribute__((ext_vector_type(8)))  float    v8f;

#define B_DIM 2048
#define I_DIM 256
#define O_DIM 256
#define KPOLY 10                 // Taylor degrees 0..9; remainder ~2e-6
#define KTOT  (KPOLY * I_DIM)    // 2560, multiple of 32
#define KHALF (KTOT / 2)         // 1280 per split-K wave

// ---------------------------------------------------------------------------
// Kernel 1: T[b][k*I + i] = (x[b,i]/4)^k   (fp16 GEMM operand, |t|<=~1.4)
// ---------------------------------------------------------------------------
__global__ void ripple_pow_kernel(const float* __restrict__ x,
                                  _Float16* __restrict__ T) {
    int id = blockIdx.x * blockDim.x + threadIdx.x;   // B*I threads
    int b = id >> 8;
    int i = id & (I_DIM - 1);
    float t = x[b * I_DIM + i] * 0.25f;
    float pw = 1.0f;
    _Float16* dst = T + (size_t)b * KTOT + i;
    #pragma unroll
    for (int k = 0; k < KPOLY; ++k) {
        dst[k * I_DIM] = (_Float16)pw;
        pw *= t;
    }
}

// ---------------------------------------------------------------------------
// Kernel 2: C[o][k*I + i] = amp * sin(p + k*pi/2)/k! * (4f)^k
//   so that  sum_k C * t^k  ==  amp * sin(f*x + p)   (Taylor about p)
// ---------------------------------------------------------------------------
__global__ void ripple_coeff_kernel(const float* __restrict__ weight,
                                    const float* __restrict__ bias,
                                    _Float16* __restrict__ Cmat) {
    int id = blockIdx.x * blockDim.x + threadIdx.x;   // O*I threads
    int o = id >> 8;
    int i = id & (I_DIM - 1);
    float amp = weight[(o * I_DIM + i) * 2 + 0];
    float f   = weight[(o * I_DIM + i) * 2 + 1];
    float p   = bias[o * (I_DIM + 1) + 1 + i];
    float s, c;
    __sincosf(p, &s, &c);
    float g = 4.0f * f;        // u = g * t
    float pw = 1.0f;           // g^k
    float invfact = 1.0f;      // 1/k!
    _Float16* dst = Cmat + (size_t)o * KTOT + i;
    #pragma unroll
    for (int k = 0; k < KPOLY; ++k) {
        float q = (k & 2) ? ((k & 1) ? -c : -s) : ((k & 1) ? c : s); // sin(p+k*pi/2)
        dst[k * I_DIM] = (_Float16)(amp * q * invfact * pw);
        pw *= g;
        invfact /= (float)(k + 1);
    }
}

// ---------------------------------------------------------------------------
// Kernel 3: WMMA GEMM, 32x32 tile per wave (2x2 blocking), split-K x2 with
// LDS reduction. Fragment addressing per CDNA5 ISA 7.12.2 (wave32, 16-bit).
// Even/odd double-buffered fragments: next-chunk loads always go to the
// opposite register set, so no WAR against in-flight WMMAs (no v_nop pads),
// and the pre-WMMA s_wait only drains the previous iteration's loads.
// ---------------------------------------------------------------------------
union FragU { uint4 u[2]; v16h v; };

#define WMMA_F16(A, B, Cacc) \
    __builtin_amdgcn_wmma_f32_16x16x32_f16(false, (A), false, (B), (short)0, \
                                           (Cacc), false, false)

__global__ __launch_bounds__(256)
void ripple_wmma_gemm_kernel(const _Float16* __restrict__ T,
                             const _Float16* __restrict__ Cmat,
                             const float* __restrict__ bias,
                             float* __restrict__ out) {
    // LDS reduction buffer: [j=0..31][tile_local=0..3][lane=0..31]
    __shared__ float red[32 * 4 * 32];

    const int lane  = threadIdx.x & 31;
    const int wave  = threadIdx.x >> 5;        // 0..7
    const int tl    = wave >> 1;               // tile within block: 0..3
    const int khalf = wave & 1;                // split-K half
    const int tid   = blockIdx.x * 4 + tl;     // 0..511 tiles (64 x 8)
    const int otile = tid & 7;
    const int btile = tid >> 3;
    const int bbase = btile * 32;
    const int obase = otile * 32;

    const int half = lane >> 4;                // 0 or 1
    const int l16  = lane & 15;
    const int kstart = khalf * KHALF;

    // Per-lane fragment base pointers (byte arithmetic, loop-invariant).
    // A lane chunks: bytes {2*k0 + half*16} and {+32}; B: {2*k0 + half*16}, {+32}.
    const char* A0p = (const char*)(T    + (size_t)(bbase +      l16) * KTOT + kstart) + half * 16;
    const char* A1p = (const char*)(T    + (size_t)(bbase + 16 + l16) * KTOT + kstart) + half * 16;
    const char* B0p = (const char*)(Cmat + (size_t)(obase +      l16) * KTOT + kstart) + half * 16;
    const char* B1p = (const char*)(Cmat + (size_t)(obase + 16 + l16) * KTOT + kstart) + half * 16;

    v8f acc00 = {}, acc01 = {}, acc10 = {}, acc11 = {};

    FragU a0[2], a1[2], b0[2], b1[2];
    a0[0].u[0] = *(const uint4*)(A0p);  a0[0].u[1] = *(const uint4*)(A0p + 32);
    a1[0].u[0] = *(const uint4*)(A1p);  a1[0].u[1] = *(const uint4*)(A1p + 32);
    b0[0].u[0] = *(const uint4*)(B0p);  b0[0].u[1] = *(const uint4*)(B0p + 32);
    b1[0].u[0] = *(const uint4*)(B1p);  b1[0].u[1] = *(const uint4*)(B1p + 32);

    constexpr int NIT = KHALF / 32;            // 40 K-steps
    #pragma unroll 2
    for (int it = 0; it < NIT; ++it) {
        const int cur = it & 1;
        const int nxt = cur ^ 1;
        const int off = (it + 1) * 64;         // next chunk; final iter lands in
                                               // guard pad (see kernel_launch)
        a0[nxt].u[0] = *(const uint4*)(A0p + off);
        a0[nxt].u[1] = *(const uint4*)(A0p + off + 32);
        a1[nxt].u[0] = *(const uint4*)(A1p + off);
        a1[nxt].u[1] = *(const uint4*)(A1p + off + 32);
        b0[nxt].u[0] = *(const uint4*)(B0p + off);
        b0[nxt].u[1] = *(const uint4*)(B0p + off + 32);
        b1[nxt].u[0] = *(const uint4*)(B1p + off);
        b1[nxt].u[1] = *(const uint4*)(B1p + off + 32);

        acc00 = WMMA_F16(a0[cur].v, b0[cur].v, acc00);   // 4 independent chains
        acc01 = WMMA_F16(a0[cur].v, b1[cur].v, acc01);
        acc10 = WMMA_F16(a1[cur].v, b0[cur].v, acc10);
        acc11 = WMMA_F16(a1[cur].v, b1[cur].v, acc11);
    }

    // ---- split-K reduction through LDS (lane-contiguous, conflict-free) ----
    if (khalf == 1) {
        #pragma unroll
        for (int r = 0; r < 8; ++r) {
            red[((0 * 8 + r) * 4 + tl) * 32 + lane] = acc00[r];
            red[((1 * 8 + r) * 4 + tl) * 32 + lane] = acc01[r];
            red[((2 * 8 + r) * 4 + tl) * 32 + lane] = acc10[r];
            red[((3 * 8 + r) * 4 + tl) * 32 + lane] = acc11[r];
        }
    }
    __syncthreads();

    if (khalf == 0) {
        // D layout: VGPR r -> M = r + 8*half (within 16-row tile); N = l16.
        const float ob0 = bias[(obase +      l16) * (I_DIM + 1)];   // bias[:,0]
        const float ob1 = bias[(obase + 16 + l16) * (I_DIM + 1)];
        #pragma unroll
        for (int r = 0; r < 8; ++r) {
            const size_t row0 = (size_t)(bbase +      half * 8 + r) * O_DIM;
            const size_t row1 = (size_t)(bbase + 16 + half * 8 + r) * O_DIM;
            out[row0 + obase +      l16] =
                acc00[r] + red[((0 * 8 + r) * 4 + tl) * 32 + lane] + ob0;
            out[row0 + obase + 16 + l16] =
                acc01[r] + red[((1 * 8 + r) * 4 + tl) * 32 + lane] + ob1;
            out[row1 + obase +      l16] =
                acc10[r] + red[((2 * 8 + r) * 4 + tl) * 32 + lane] + ob0;
            out[row1 + obase + 16 + l16] =
                acc11[r] + red[((3 * 8 + r) * 4 + tl) * 32 + lane] + ob1;
        }
    }
}

// ---------------------------------------------------------------------------
extern "C" void kernel_launch(void* const* d_in, const int* in_sizes, int n_in,
                              void* d_out, int out_size, void* d_ws, size_t ws_size,
                              hipStream_t stream) {
    const float* x      = (const float*)d_in[0];   // [2048, 256]
    const float* weight = (const float*)d_in[1];   // [256, 256, 2]
    const float* bias   = (const float*)d_in[2];   // [256, 257]
    float* out = (float*)d_out;                    // [2048, 256]

    // Workspace layout: T (10.5 MB) | Cmat (1.3 MB) | 4 KB guard pad for the
    // double-buffer's one-past-end prefetch loads on the final K iteration.
    _Float16* T    = (_Float16*)d_ws;                      // 2048*2560 f16
    _Float16* Cmat = T + (size_t)B_DIM * KTOT;             // 256*2560  f16

    ripple_pow_kernel<<<(B_DIM * I_DIM) / 256, 256, 0, stream>>>(x, T);
    ripple_coeff_kernel<<<(O_DIM * I_DIM) / 256, 256, 0, stream>>>(weight, bias, Cmat);

    // 512 tiles (32x32) x 2 split-K waves = 1024 waves; 8 waves/block -> 128 blocks
    ripple_wmma_gemm_kernel<<<128, 256, 0, stream>>>(T, Cmat, bias, out);
}
